// H_WordAttNet_37838661878285
// MI455X (gfx1250) — matrix operand order repaired and checked by reading, not compiled, not run on hardware
//
#include <hip/hip_runtime.h>
#include <hip/hip_bf16.h>

// Problem constants (reference: VOCAB=100000, EMB=300, T=128, B=512, H=128, D=2H=256)
#define T_    128
#define B_    512
#define H_    128
#define D_    256
#define EMB_  300
#define EMBP  320          // EMB padded to multiple of 32 (K of bf16 WMMA)
#define G3    384          // 3*H gate columns
#define EPS_  1e-7f

typedef __attribute__((ext_vector_type(16))) __bf16 v16bf;
typedef __attribute__((ext_vector_type(8)))  float  v8f;

__device__ __forceinline__ v8f wmma_bf16(v16bf a, v16bf b, v8f c) {
  // D = A(16x32 bf16) * B(32x16 bf16) + C(16x16 f32)
  return __builtin_amdgcn_wmma_f32_16x16x32_bf16(false, a, false, b, (short)0, c,
                                                 false, false);
}

// ---------------------------------------------------------------------------
// Weight conversion f32 -> bf16 with column padding
// ---------------------------------------------------------------------------
__global__ void k_cvt_pad(const float* __restrict__ src, __bf16* __restrict__ dst,
                          int rows, int scols, int dcols) {
  int i = blockIdx.x * blockDim.x + threadIdx.x;
  if (i >= rows * dcols) return;
  int r = i / dcols, c = i - r * dcols;
  dst[i] = (c < scols) ? (__bf16)src[r * scols + c] : (__bf16)0.0f;
}

__global__ void k_cvt(const float* __restrict__ src, __bf16* __restrict__ dst, int n) {
  int i = blockIdx.x * blockDim.x + threadIdx.x;
  if (i < n) dst[i] = (__bf16)src[i];
}

// ---------------------------------------------------------------------------
// Fused embedding-gather + input-projection GEMM (both directions).
// C[m, n] = emb[tokens[m]] . W_ih[dir][g] + b_ih ; m = t*B + b, n in [0,768)
// Grid: (M/16, 6), 256 threads = 8 waves; wave w owns cols ngrp*128 + w*16.
// dir1 result stored time-reversed (x[::-1] of the reference).
// ---------------------------------------------------------------------------
__global__ __launch_bounds__(256)
void k_ip_gemm(const int* __restrict__ tokens,
               const float* __restrict__ emb,
               const __bf16* __restrict__ Wih,   // [2][384][EMBP] bf16
               const float* __restrict__ bih_f,
               const float* __restrict__ bih_b,
               float* __restrict__ xp)           // [2][T][B][384] f32
{
  __shared__ __bf16 At[16][EMBP];
  __shared__ int s_tok[16];
  const int mtile = blockIdx.x;        // 0..4095
  const int ngrp  = blockIdx.y;        // 0..5
  const int tid   = threadIdx.x;
  const int m0 = mtile * 16;
  const int t  = m0 / B_;              // tiles never straddle t (512 % 16 == 0)
  const int b0 = m0 - t * B_;

  if (tid < 16) s_tok[tid] = tokens[t * B_ + b0 + tid];
  __syncthreads();
  for (int i = tid; i < 16 * EMBP; i += 256) {
    int r = i / EMBP, c = i - r * EMBP;
    At[r][c] = (c < EMB_) ? (__bf16)emb[(long)s_tok[r] * EMB_ + c] : (__bf16)0.0f;
  }
  __syncthreads();

  const int wave  = tid >> 5;
  const int lane  = tid & 31;
  const int khalf = lane >> 4;         // 0: lanes 0-15, 1: lanes 16-31
  const int nc  = ngrp * 128 + wave * 16 + (lane & 15);   // global col [0,768)
  const int dir = (nc >= G3) ? 1 : 0;
  const int g   = nc - dir * G3;

  v8f acc = {};
  const __bf16* wb_base = Wih + ((long)dir * G3 + g) * EMBP;
#pragma unroll
  for (int ks = 0; ks < EMBP / 32; ++ks) {
    v16bf a, bb;
    const __bf16* pa = &At[lane & 15][ks * 32 + khalf * 16];
#pragma unroll
    for (int e = 0; e < 8; ++e) { a[e] = pa[e]; a[e + 8] = pa[8 + e]; }
    const __bf16* pb = wb_base + ks * 32 + khalf * 16;
#pragma unroll
    for (int e = 0; e < 16; ++e) bb[e] = pb[e];
    acc = wmma_bf16(a, bb, acc);
  }

  const float bias = dir ? bih_b[g] : bih_f[g];
  const int tt = dir ? (T_ - 1 - t) : t;
  const int rbase = khalf * 8;
#pragma unroll
  for (int v = 0; v < 8; ++v) {
    int bi = b0 + rbase + v;
    xp[(((long)dir * T_ + tt) * B_ + bi) * G3 + g] = acc[v] + bias;
  }
}

// ---------------------------------------------------------------------------
// GRU recurrence. 64 WGs: dir = blk>>5, batch tile of 16 rows = blk&31.
// W_hh B-fragments stay resident in VGPRs for all 128 steps; h lives in LDS.
// ---------------------------------------------------------------------------
__global__ __launch_bounds__(256)
void k_gru(const float* __restrict__ h0,       // [2][B][H]
           const __bf16* __restrict__ Whh,     // [2][384][128] bf16
           const float* __restrict__ bhh_f,
           const float* __restrict__ bhh_b,
           const float* __restrict__ xp,       // [2][T][B][384]
           float* __restrict__ fout,           // [T][B][256]
           float* __restrict__ hlast)          // [2][B][H]
{
  __shared__ float sh[16][H_];    // running hidden state (f32)
  __shared__ float sg[16][G3];    // per-step gate pre-activations
  __shared__ float sbh[G3];

  const int dir   = blockIdx.x >> 5;
  const int btile = blockIdx.x & 31;
  const int bg0   = btile * 16;
  const int tid   = threadIdx.x;
  const int wave  = tid >> 5;
  const int lane  = tid & 31;
  const int khalf = lane >> 4;

  const float* bhh = dir ? bhh_b : bhh_f;
  for (int i = tid; i < G3; i += 256) sbh[i] = bhh[i];
  for (int i = tid; i < 16 * H_; i += 256) {
    int r = i >> 7, k = i & (H_ - 1);
    sh[r][k] = h0[((long)dir * B_ + bg0 + r) * H_ + k];
  }

  // Preload constant W_hh^T fragments: wave owns 3 N-tiles (48 cols), K=128.
  v16bf bf[3][4];
#pragma unroll
  for (int ti = 0; ti < 3; ++ti) {
    int g = wave * 48 + ti * 16 + (lane & 15);
    const __bf16* wb = Whh + ((long)dir * G3 + g) * H_;
#pragma unroll
    for (int ks = 0; ks < 4; ++ks) {
      const __bf16* p = wb + ks * 32 + khalf * 16;
#pragma unroll
      for (int e = 0; e < 16; ++e) bf[ti][ks][e] = p[e];
    }
  }
  __syncthreads();

  const long xpd = (long)dir * T_ * B_ * G3;
  for (int t = 0; t < T_; ++t) {
    // A fragments from current h (f32 LDS -> bf16)
    v16bf af[4];
    {
      int r = lane & 15;
#pragma unroll
      for (int ks = 0; ks < 4; ++ks) {
        const float* ph = &sh[r][ks * 32 + khalf * 16];
#pragma unroll
        for (int e = 0; e < 8; ++e) {
          af[ks][e]     = (__bf16)ph[e];
          af[ks][e + 8] = (__bf16)ph[8 + e];
        }
      }
    }
    const float* xpt = xp + xpd + (long)t * B_ * G3;

#pragma unroll
    for (int ti = 0; ti < 3; ++ti) {
      v8f c = {};
#pragma unroll
      for (int ks = 0; ks < 4; ++ks) c = wmma_bf16(af[ks], bf[ti][ks], c);
      int g = wave * 48 + ti * 16 + (lane & 15);
      int rbase = khalf * 8;
#pragma unroll
      for (int v = 0; v < 8; ++v) {
        int row = rbase + v;
        float val = c[v] + sbh[g];
        // r,z gates: add x-projection now. n gate: keep hn raw (needs r*hn).
        if (g < 2 * H_) val += xpt[(long)(bg0 + row) * G3 + g];
        sg[row][g] = val;
      }
    }
    __syncthreads();

    // Gate nonlinearity: 2048 elements over 256 threads
#pragma unroll
    for (int i = 0; i < 8; ++i) {
      int e = i * 256 + tid;
      int r = e >> 7, j = e & (H_ - 1);
      int bgl = bg0 + r;
      float rg = 1.f / (1.f + __expf(-sg[r][j]));
      float zg = 1.f / (1.f + __expf(-sg[r][j + H_]));
      float hn = sg[r][j + 2 * H_];
      float xn = xpt[(long)bgl * G3 + 2 * H_ + j];
      float nn = tanhf(xn + rg * hn);
      float hnew = (1.f - zg) * nn + zg * sh[r][j];
      sh[r][j] = hnew;
      long fo = dir ? (((long)(T_ - 1 - t) * B_ + bgl) * D_ + H_ + j)
                    : (((long)t * B_ + bgl) * D_ + j);
      fout[fo] = hnew;
      if (t == T_ - 1) hlast[((long)dir * B_ + bgl) * H_ + j] = hnew;
    }
    __syncthreads();
  }
}

// ---------------------------------------------------------------------------
// hyp = tanh(f_output @ attn2_W^T + attn2_b). Pure WMMA, no LDS needed:
// A fragments load 16B-contiguous bf16 runs straight from global.
// ---------------------------------------------------------------------------
__global__ __launch_bounds__(256)
void k_attn_gemm(const __bf16* __restrict__ fbf,   // [T*B][256] bf16
                 const __bf16* __restrict__ Wat,   // [256][256] bf16 (row=N, col=K)
                 const float* __restrict__ ab,     // attn2_b [256]
                 float* __restrict__ hyp)          // [T*B][256] f32
{
  const int tile  = blockIdx.x * 8 + (threadIdx.x >> 5);
  const int lane  = threadIdx.x & 31;
  const int mtile = tile >> 4;
  const int ntile = tile & 15;
  const int khalf = lane >> 4;
  const int m  = mtile * 16 + (lane & 15);
  const int nc = ntile * 16 + (lane & 15);

  v8f acc = {};
  const __bf16* pa_base = fbf + (long)m * D_;
  const __bf16* pb_base = Wat + (long)nc * D_;
#pragma unroll
  for (int ks = 0; ks < D_ / 32; ++ks) {
    v16bf a, bb;
    const __bf16* pa = pa_base + ks * 32 + khalf * 16;
#pragma unroll
    for (int e = 0; e < 8; ++e) { a[e] = pa[e]; a[e + 8] = pa[8 + e]; }
    const __bf16* pb = pb_base + ks * 32 + khalf * 16;
#pragma unroll
    for (int e = 0; e < 16; ++e) bb[e] = pb[e];
    acc = wmma_bf16(a, bb, acc);
  }
  const float bias = ab[nc];
  const int m0 = mtile * 16, rbase = khalf * 8;
#pragma unroll
  for (int v = 0; v < 8; ++v)
    hyp[(long)(m0 + rbase + v) * D_ + nc] = tanhf(acc[v] + bias);
}

// ---------------------------------------------------------------------------
// Hyperbolic attention scores + softmax over T + Klein/Einstein midpoint.
// One block per batch column b; thread d owns feature dim d (D_ == 256).
// ---------------------------------------------------------------------------
__device__ __forceinline__ float block_reduce(float v, float* s, int tid) {
  __syncthreads();
  s[tid] = v;
  __syncthreads();
  for (int off = 128; off > 0; off >>= 1) {
    if (tid < off) s[tid] += s[tid + off];
    __syncthreads();
  }
  float r = s[0];
  __syncthreads();
  return r;
}

__global__ __launch_bounds__(256)
void k_pool(const float* __restrict__ fout,  // [T][B][256]
            const float* __restrict__ hyp,   // [T][B][256]
            const float* __restrict__ centroid,
            const float* __restrict__ beta_p,
            float* __restrict__ out)         // [B][256]
{
  __shared__ float s_red[256];
  __shared__ float s_sc[T_], s_gam[T_], s_kf[T_], s_cf[T_];
  const int b = blockIdx.x;
  const int d = threadIdx.x;
  const float beta = beta_p[0];

  // Lorentz lift of centroid (shared scalars computed redundantly per thread)
  float cd = centroid[d];
  float rc = sqrtf(block_reduce(cd * cd, s_red, d));
  float chat = cd / rc;
  float shc = sinhf(rc), chc = coshf(rc);

  for (int t = 0; t < T_; ++t) {
    long base = ((long)t * B_ + b) * D_;
    float f = fout[base + d];
    float h = hyp[base + d];
    float sf2 = block_reduce(f * f, s_red, d);
    float sh2 = block_reduce(h * h, s_red, d);
    float dhc = block_reduce(h * chat, s_red, d);
    if (d == 0) {
      float r1 = sqrtf(sf2);
      float r2 = sqrtf(sh2);
      // Minkowski inner product with centroid lift
      float pm = -(dhc / r2) * sinhf(r2) * shc + coshf(r2) * chc;
      pm = fminf(fmaxf(pm, 1.f + EPS_), 1e16f);
      float dist = logf(pm) + log1pf(sqrtf(pm * pm - 1.f + EPS_) / pm);
      s_sc[t] = -beta * dist - 1.f;
      float th = tanhf(r1);               // ||klein|| = tanh(||f||)
      float gsq = fminf(fmaxf(1.f - th * th, EPS_), 1.f - EPS_);
      float gam = 1.f / sqrtf(gsq);
      s_gam[t] = fminf(fmaxf(gam, 1.f + EPS_), 1e16f);
      s_kf[t] = th / r1;                  // klein = (tanh r / r) * f
    }
  }
  __syncthreads();

  // softmax over T (threads 0..127 each own a timestep)
  float sc = (d < T_) ? s_sc[d] : -3.0e38f;
  s_red[d] = sc;
  __syncthreads();
  for (int off = 128; off > 0; off >>= 1) {
    if (d < off) s_red[d] = fmaxf(s_red[d], s_red[d + off]);
    __syncthreads();
  }
  float mx = s_red[0];
  __syncthreads();
  float al = (d < T_) ? __expf(sc - mx) : 0.f;
  float sa = block_reduce(al, s_red, d);
  float w  = (d < T_) ? (al / sa) * s_gam[d] : 0.f;
  float sw = block_reduce(w, s_red, d);
  if (d < T_) s_cf[d] = (w / sw) * s_kf[d];
  __syncthreads();

  float accv = 0.f;
  for (int t = 0; t < T_; ++t)
    accv += s_cf[t] * fout[((long)t * B_ + b) * D_ + d];
  out[(long)b * D_ + d] = accv;
}

// ---------------------------------------------------------------------------
extern "C" void kernel_launch(void* const* d_in, const int* in_sizes, int n_in,
                              void* d_out, int out_size, void* d_ws, size_t ws_size,
                              hipStream_t stream)
{
  (void)in_sizes; (void)n_in; (void)out_size; (void)ws_size;
  const int*   tokens = (const int*)d_in[0];
  const float* h0     = (const float*)d_in[1];
  const float* emb    = (const float*)d_in[2];
  const float* Wih_f  = (const float*)d_in[3];
  const float* Whh_f  = (const float*)d_in[4];
  const float* bih_f  = (const float*)d_in[5];
  const float* bhh_f  = (const float*)d_in[6];
  const float* Wih_b  = (const float*)d_in[7];
  const float* Whh_b  = (const float*)d_in[8];
  const float* bih_b  = (const float*)d_in[9];
  const float* bhh_b  = (const float*)d_in[10];
  const float* attW   = (const float*)d_in[11];
  const float* attb   = (const float*)d_in[12];
  const float* cent   = (const float*)d_in[13];
  const float* beta   = (const float*)d_in[14];

  float* out   = (float*)d_out;          // [1,B,D] first
  float* hlast = out + (size_t)B_ * D_;  // then [2,B,H]

  // workspace carve-out
  char* ws = (char*)d_ws;
  size_t o = 0;
  __bf16* Wih_bf = (__bf16*)(ws + o); o += (size_t)2 * G3 * EMBP * 2;
  __bf16* Whh_bf = (__bf16*)(ws + o); o += (size_t)2 * G3 * H_ * 2;
  __bf16* Wat_bf = (__bf16*)(ws + o); o += (size_t)D_ * D_ * 2;
  o = (o + 255) & ~(size_t)255;
  float*  xp     = (float*)(ws + o);  o += (size_t)2 * T_ * B_ * G3 * 4;
  float*  fout   = (float*)(ws + o);  o += (size_t)T_ * B_ * D_ * 4;
  float*  hyp    = (float*)(ws + o);  o += (size_t)T_ * B_ * D_ * 4;
  __bf16* fbf    = (__bf16*)(ws + o); o += (size_t)T_ * B_ * D_ * 2;

  // 1) weight precision conversion
  {
    int n1 = G3 * EMBP;
    k_cvt_pad<<<(n1 + 255) / 256, 256, 0, stream>>>(Wih_f, Wih_bf, G3, EMB_, EMBP);
    k_cvt_pad<<<(n1 + 255) / 256, 256, 0, stream>>>(Wih_b, Wih_bf + (size_t)G3 * EMBP,
                                                    G3, EMB_, EMBP);
    int n2 = G3 * H_;
    k_cvt_pad<<<(n2 + 255) / 256, 256, 0, stream>>>(Whh_f, Whh_bf, G3, H_, H_);
    k_cvt_pad<<<(n2 + 255) / 256, 256, 0, stream>>>(Whh_b, Whh_bf + (size_t)G3 * H_,
                                                    G3, H_, H_);
    int n3 = D_ * D_;
    k_cvt_pad<<<(n3 + 255) / 256, 256, 0, stream>>>(attW, Wat_bf, D_, D_, D_);
  }

  // 2) fused gather + input projections (both dirs), WMMA bf16
  dim3 gip(T_ * B_ / 16, 6);
  k_ip_gemm<<<gip, 256, 0, stream>>>(tokens, emb, Wih_bf, bih_f, bih_b, xp);

  // 3) bidirectional GRU recurrence (register-resident W_hh fragments)
  k_gru<<<64, 256, 0, stream>>>(h0, Whh_bf, bhh_f, bhh_b, xp, fout, hlast);

  // 4) f_output -> bf16 for attention GEMM
  int nf = T_ * B_ * D_;
  k_cvt<<<(nf + 255) / 256, 256, 0, stream>>>(fout, fbf, nf);

  // 5) attention projection + tanh, WMMA bf16
  k_attn_gemm<<<(T_ * B_ / 16) * (D_ / 16) / 8, 256, 0, stream>>>(fbf, Wat_bf, attb, hyp);

  // 6) hyperbolic scores, softmax over T, Einstein-midpoint pooling
  k_pool<<<B_, 256, 0, stream>>>(fout, hyp, cent, beta, out);
}
